// CrossModalAlignment_14010183320212
// MI455X (gfx1250) — compile-verified, matrix operand order checked by reading, not run on hardware
//
#include <hip/hip_runtime.h>

// ---------------------------------------------------------------------------
// CrossModalAlignment for gfx1250 (MI455X): WMMA f16->f32 GEMM pipeline.
// v2: register-level N-blocking (16x64 per wave) for 4x A-operand reuse.
// ---------------------------------------------------------------------------

typedef __attribute__((ext_vector_type(16))) _Float16 v16h;
typedef __attribute__((ext_vector_type(8)))  _Float16 v8h;
typedef __attribute__((ext_vector_type(8)))  float    v8f;

#define BB 4      // batch
#define NN 256    // tokens
#define DD 128    // model dim
#define NH 4      // heads
#define DH 32     // head dim
#define NREL 4

union HV { v16h v; v8h h[2]; };

__device__ __forceinline__ v16h frag_load(const _Float16* p) {
  HV u;
  u.h[0] = *(const v8h*)(p);
  u.h[1] = *(const v8h*)(p + 16);
  return u.v;
}

// ---------------------------------------------------------------------------
// Generic WMMA GEMM: C[z] = act(A[z] (MxK) * W[z]^T (NxK) + bias), f16 inputs.
// Per-z offsets: z = zo*inner + zi -> base + zo*s?o + zi*s?i.
// Output row remap: row = rowMul*m + rowOff  (for strided conv-transpose rows).
// One wave (32 threads) per 16 x (16*NB) output tile; A frag reused NB times.
// ---------------------------------------------------------------------------
template <int NB>
__global__ __launch_bounds__(32)
void cma_gemm(const _Float16* __restrict__ A, int lda, long sAo, long sAi,
              const _Float16* __restrict__ W, int ldw, long sWo, long sWi,
              float* __restrict__ C32, int ldc32, long sC32o, long sC32i,
              _Float16* __restrict__ C16, int ldc16, long sC16o, long sC16i,
              const float* __restrict__ bias,
              int K, int inner, int relu, int rowMul, int rowOff)
{
  const int lane = threadIdx.x;
  const int l16 = lane & 15, hi = lane >> 4;
  const int z = blockIdx.z;
  const int zo = z / inner, zi = z - zo * inner;
  const int m0 = blockIdx.y * 16, n0 = blockIdx.x * (16 * NB);

  const _Float16* Ab = A + zo * sAo + zi * sAi + (size_t)(m0 + l16) * lda + 8 * hi;
  const _Float16* Wb[NB];
#pragma unroll
  for (int t = 0; t < NB; ++t)
    Wb[t] = W + zo * sWo + zi * sWi + (size_t)(n0 + t * 16 + l16) * ldw + 8 * hi;

  v8f acc[NB];
#pragma unroll
  for (int t = 0; t < NB; ++t) acc[t] = (v8f){0.f, 0.f, 0.f, 0.f, 0.f, 0.f, 0.f, 0.f};

  for (int kb = 0; kb < K; kb += 32) {
    v16h af = frag_load(Ab + kb);
#pragma unroll
    for (int t = 0; t < NB; ++t) {
      v16h wf = frag_load(Wb[t] + kb);
      acc[t] = __builtin_amdgcn_wmma_f32_16x16x32_f16(false, af, false, wf,
                                                      (short)0, acc[t], false, false);
    }
  }

#pragma unroll
  for (int t = 0; t < NB; ++t) {
    const int n = n0 + t * 16 + l16;
    const float bn = bias ? bias[n] : 0.f;
#pragma unroll
    for (int r = 0; r < 8; ++r) {
      int m = m0 + hi * 8 + r;
      float v = acc[t][r] + bn;
      if (relu) v = v > 0.f ? v : 0.f;
      long row = (long)rowMul * m + rowOff;
      if (C32) C32[zo * sC32o + zi * sC32i + row * (long)ldc32 + n] = v;
      if (C16) C16[zo * sC16o + zi * sC16i + row * (long)ldc16 + n] = (_Float16)v;
    }
  }
}

// ---------------------------------------------------------------------------
// Fused relational block: for one (b,i,r):
//   h2 = relu(relu(a_i (+) c) @ w2^T + b2)   (256x64, WMMA)
//   s_j = h2[j].w3 + b3 ; p = softmax(s) ; feat = p^T f2  -> allr[:, r*128..]
// 256 threads = 8 waves; wave w owns j-tiles {2w,2w+1}; the relu(a+c)
// fragment is built once per k-step and reused across the 4 n-tiles.
// ---------------------------------------------------------------------------
__global__ __launch_bounds__(256)
void cma_rel(const _Float16* __restrict__ a16,   // [NREL][BB*NN][DD]
             const _Float16* __restrict__ c16,   // [NREL][BB*NN][DD]
             const _Float16* __restrict__ f2,    // [BB*NN][DD]
             const _Float16* __restrict__ w2all, // [NREL][64][128]
             const float* __restrict__ b2all,    // [NREL][64]
             const float* __restrict__ w3all,    // [NREL][64]
             const float* __restrict__ b3all,    // [NREL]
             _Float16* __restrict__ allr)        // [BB*NN][NREL*DD]
{
  const int bi = blockIdx.x;           // b*NN + i
  const int b  = bi >> 8;
  const int r  = blockIdx.y;
  const int tid = threadIdx.x;
  const int wave = tid >> 5, lane = tid & 31;
  const int l16 = lane & 15, hi = lane >> 4;

  __shared__ float sc[NN];
  __shared__ float red[8];

  const _Float16* arow  = a16 + ((size_t)r * BB * NN + bi) * DD;
  const _Float16* cbase = c16 + ((size_t)r * BB * NN + (size_t)b * NN) * DD;
  const _Float16* w2    = w2all + (size_t)r * 64 * 128;
  const float* b2 = b2all + r * 64;
  const float* w3 = w3all + r * 64;

  for (int jt = wave * 2; jt < wave * 2 + 2; ++jt) {
    const _Float16* cp = cbase + (size_t)(jt * 16 + l16) * DD + 8 * hi;
    const _Float16* ap = arow + 8 * hi;
    const _Float16* wp = w2 + (size_t)l16 * 128 + 8 * hi;

    v8f acc[4];
#pragma unroll
    for (int nt = 0; nt < 4; ++nt)
      acc[nt] = (v8f){0.f, 0.f, 0.f, 0.f, 0.f, 0.f, 0.f, 0.f};

    for (int kb = 0; kb < 128; kb += 32) {
      HV uc, ua;
      uc.h[0] = *(const v8h*)(cp + kb);
      uc.h[1] = *(const v8h*)(cp + kb + 16);
      ua.h[0] = *(const v8h*)(ap + kb);
      ua.h[1] = *(const v8h*)(ap + kb + 16);
      v16h hf;
#pragma unroll
      for (int e = 0; e < 16; ++e) {
        _Float16 s = uc.v[e] + ua.v[e];
        hf[e] = s > (_Float16)0 ? s : (_Float16)0;
      }
#pragma unroll
      for (int nt = 0; nt < 4; ++nt) {
        v16h wf = frag_load(wp + (size_t)nt * 16 * 128 + kb);
        acc[nt] = __builtin_amdgcn_wmma_f32_16x16x32_f16(false, hf, false, wf,
                                                         (short)0, acc[nt], false, false);
      }
    }

    float sAcc[8] = {0.f, 0.f, 0.f, 0.f, 0.f, 0.f, 0.f, 0.f};
#pragma unroll
    for (int nt = 0; nt < 4; ++nt) {
      const int n = nt * 16 + l16;
      const float w3n = w3[n], b2n = b2[n];
#pragma unroll
      for (int q = 0; q < 8; ++q) {
        float v = acc[nt][q] + b2n;
        v = v > 0.f ? v : 0.f;
        sAcc[q] += v * w3n;
      }
    }
#pragma unroll
    for (int q = 0; q < 8; ++q) {
      float v = sAcc[q];
      v += __shfl_xor(v, 1, 32);
      v += __shfl_xor(v, 2, 32);
      v += __shfl_xor(v, 4, 32);
      v += __shfl_xor(v, 8, 32);
      sAcc[q] = v;
    }
    if (l16 == 0) {
      float b3v = b3all[r];
#pragma unroll
      for (int q = 0; q < 8; ++q)
        sc[jt * 16 + hi * 8 + q] = sAcc[q] + b3v;
    }
  }
  __syncthreads();

  // softmax over 256 scores
  float v = sc[tid];
  float m = v;
  for (int off = 16; off > 0; off >>= 1) m = fmaxf(m, __shfl_xor(m, off, 32));
  if (lane == 0) red[wave] = m;
  __syncthreads();
  float bm = red[0];
#pragma unroll
  for (int w = 1; w < 8; ++w) bm = fmaxf(bm, red[w]);
  float e = expf(v - bm);
  float s = e;
  for (int off = 16; off > 0; off >>= 1) s += __shfl_xor(s, off, 32);
  __syncthreads();
  if (lane == 0) red[wave] = s;
  __syncthreads();
  float tot = 0.f;
#pragma unroll
  for (int w = 0; w < 8; ++w) tot += red[w];
  sc[tid] = e / tot;
  __syncthreads();

  // feat[d] = sum_j p_j * f2[b, j, d]
  if (tid < DD) {
    const _Float16* f2b = f2 + (size_t)b * NN * DD + tid;
    float acc2 = 0.f;
    for (int j = 0; j < NN; ++j) acc2 += sc[j] * (float)f2b[(size_t)j * DD];
    allr[(size_t)bi * (NREL * DD) + r * DD + tid] = (_Float16)acc2;
  }
}

// --------------------------- elementwise kernels ---------------------------

__global__ void cma_ln(const float* __restrict__ x, const float* __restrict__ g,
                       const float* __restrict__ be, float* __restrict__ y32,
                       _Float16* __restrict__ y16, int D)
{
  const int row = blockIdx.x, tid = threadIdx.x;
  __shared__ float red[4];
  const float* xr = x + (size_t)row * D;
  float acc = 0.f;
  for (int d = tid; d < D; d += 128) acc += xr[d];
  for (int off = 16; off > 0; off >>= 1) acc += __shfl_xor(acc, off, 32);
  if ((tid & 31) == 0) red[tid >> 5] = acc;
  __syncthreads();
  float mu = (red[0] + red[1] + red[2] + red[3]) / (float)D;
  __syncthreads();
  float va = 0.f;
  for (int d = tid; d < D; d += 128) { float t = xr[d] - mu; va += t * t; }
  for (int off = 16; off > 0; off >>= 1) va += __shfl_xor(va, off, 32);
  if ((tid & 31) == 0) red[tid >> 5] = va;
  __syncthreads();
  float rstd = rsqrtf((red[0] + red[1] + red[2] + red[3]) / (float)D + 1e-5f);
  for (int d = tid; d < D; d += 128) {
    float o = (xr[d] - mu) * rstd * g[d] + be[d];
    if (y32) y32[(size_t)row * D + d] = o;
    if (y16) y16[(size_t)row * D + d] = (_Float16)o;
  }
}

__global__ void cma_softmax(const float* __restrict__ s, _Float16* __restrict__ p,
                            int L, float scale)
{
  const int row = blockIdx.x, tid = threadIdx.x;
  __shared__ float red[8];
  const int nw = blockDim.x >> 5;
  float v = s[(size_t)row * L + tid] * scale;
  float m = v;
  for (int off = 16; off > 0; off >>= 1) m = fmaxf(m, __shfl_xor(m, off, 32));
  if ((tid & 31) == 0) red[tid >> 5] = m;
  __syncthreads();
  float bm = -3.4e38f;
  for (int i = 0; i < nw; ++i) bm = fmaxf(bm, red[i]);
  float e = expf(v - bm);
  float sm = e;
  for (int off = 16; off > 0; off >>= 1) sm += __shfl_xor(sm, off, 32);
  __syncthreads();
  if ((tid & 31) == 0) red[tid >> 5] = sm;
  __syncthreads();
  float tot = 0.f;
  for (int i = 0; i < nw; ++i) tot += red[i];
  p[(size_t)row * L + tid] = (_Float16)(e / tot);
}

__global__ void cma_cvt(const float* __restrict__ x, _Float16* __restrict__ y, int n)
{
  int i = blockIdx.x * blockDim.x + threadIdx.x;
  if (i < n) y[i] = (_Float16)x[i];
}

// conv1d k3 s2 p1 im2col: x16 (BB,S,128) -> col (BB,S/2,384)
__global__ void cma_im2col_down(const _Float16* __restrict__ x,
                                _Float16* __restrict__ col, int S)
{
  int idx = blockIdx.x * blockDim.x + threadIdx.x;
  int S2 = S >> 1;
  int total = BB * S2 * 384;
  if (idx >= total) return;
  int c = idx % 384;
  int t = (idx / 384) % S2;
  int b = idx / (384 * S2);
  int kk = c >> 7, i = c & 127;
  int j = 2 * t - 1 + kk;
  _Float16 v = (_Float16)0;
  if (j >= 0 && j < S) v = x[((size_t)b * S + j) * DD + i];
  col[idx] = v;
}

// convT k4 s2 p1 im2col (one parity): x16 (BB,S,128) -> col (BB,S,256)
__global__ void cma_im2col_up(const _Float16* __restrict__ x,
                              _Float16* __restrict__ col, int S, int o0, int o1)
{
  int idx = blockIdx.x * blockDim.x + threadIdx.x;
  int total = BB * S * 256;
  if (idx >= total) return;
  int c = idx % 256;
  int s = (idx / 256) % S;
  int b = idx / (256 * S);
  int tap = c >> 7, i = c & 127;
  int j = s + (tap == 0 ? o0 : o1);
  _Float16 v = (_Float16)0;
  if (j >= 0 && j < S) v = x[((size_t)b * S + j) * DD + i];
  col[idx] = v;
}

// V16 (BB,Nk,128) -> vT (BB,NH,DH,Nk)
__global__ void cma_vT(const _Float16* __restrict__ V, _Float16* __restrict__ vT, int Nk)
{
  int idx = blockIdx.x * blockDim.x + threadIdx.x;
  int total = BB * NH * DH * Nk;
  if (idx >= total) return;
  int k = idx % Nk;
  int d = (idx / Nk) % DH;
  int h = (idx / (Nk * DH)) % NH;
  int b = idx / (Nk * DH * NH);
  vT[idx] = V[((size_t)b * Nk + k) * DD + h * DH + d];
}

// down_w (128,128,3) f32 -> dwT (128, 3*128) f16, k-major per tap
__global__ void cma_pack_dw(const float* __restrict__ w, _Float16* __restrict__ o)
{
  int idx = blockIdx.x * blockDim.x + threadIdx.x;
  if (idx >= 128 * 384) return;
  int c = idx % 384, oo = idx / 384;
  int kk = c >> 7, i = c & 127;
  o[idx] = (_Float16)w[((size_t)oo * 128 + i) * 3 + kk];
}

// up_w (Cin=128, Cout=128, 4) f32 -> (128, 256) f16 with taps kk0,kk1
__global__ void cma_pack_uw(const float* __restrict__ w, _Float16* __restrict__ o,
                            int kk0, int kk1)
{
  int idx = blockIdx.x * blockDim.x + threadIdx.x;
  if (idx >= 128 * 256) return;
  int c = idx % 256, oo = idx / 256;
  int tap = c >> 7, i = c & 127;
  int kk = tap ? kk1 : kk0;
  o[idx] = (_Float16)w[((size_t)i * 128 + oo) * 4 + kk];
}

// combined[:,0:128] = softmax(lw) . {a0, a1u, a2u}
__global__ void cma_wsum(const float* __restrict__ a0, const float* __restrict__ a1u,
                         const float* __restrict__ a2u, const float* __restrict__ lw,
                         _Float16* __restrict__ comb)
{
  int idx = blockIdx.x * blockDim.x + threadIdx.x;
  int total = BB * NN * DD;
  if (idx >= total) return;
  int d = idx % DD;
  long row = idx / DD;
  float w0 = lw[0], w1 = lw[1], w2 = lw[2];
  float mx = fmaxf(w0, fmaxf(w1, w2));
  float e0 = expf(w0 - mx), e1 = expf(w1 - mx), e2 = expf(w2 - mx);
  float inv = 1.f / (e0 + e1 + e2);
  float v = (e0 * a0[idx] + e1 * a1u[idx] + e2 * a2u[idx]) * inv;
  comb[row * 384 + d] = (_Float16)v;
}

__global__ void cma_resid(float* __restrict__ al, const float* __restrict__ proj,
                          _Float16* __restrict__ m16, int n)
{
  int i = blockIdx.x * blockDim.x + threadIdx.x;
  if (i >= n) return;
  float v = al[i] + 0.1f * proj[i];
  al[i] = v;
  m16[i] = (_Float16)v;
}

// --------------------------- host-side helpers -----------------------------

struct Arena {
  char* p; size_t off;
  void* take(size_t bytes) {
    size_t o = (off + 255) & ~(size_t)255;
    off = o + bytes;
    return p + o;
  }
};

static inline void gemm(hipStream_t st,
                        const _Float16* A, int lda, long sAo, long sAi,
                        const _Float16* W, int ldw, long sWo, long sWi,
                        const float* bias, int relu,
                        float* C32, int ldc32, long sC32o, long sC32i,
                        _Float16* C16, int ldc16, long sC16o, long sC16i,
                        int M, int N, int K, int outer, int inner,
                        int rowMul = 1, int rowOff = 0)
{
  int nb = (N % 64 == 0) ? 4 : (N % 32 == 0) ? 2 : 1;
  dim3 g(N / (16 * nb), M / 16, outer * inner);
  if (nb == 4)
    cma_gemm<4><<<g, 32, 0, st>>>(A, lda, sAo, sAi, W, ldw, sWo, sWi,
                                  C32, ldc32, sC32o, sC32i, C16, ldc16, sC16o, sC16i,
                                  bias, K, inner, relu, rowMul, rowOff);
  else if (nb == 2)
    cma_gemm<2><<<g, 32, 0, st>>>(A, lda, sAo, sAi, W, ldw, sWo, sWi,
                                  C32, ldc32, sC32o, sC32i, C16, ldc16, sC16o, sC16i,
                                  bias, K, inner, relu, rowMul, rowOff);
  else
    cma_gemm<1><<<g, 32, 0, st>>>(A, lda, sAo, sAi, W, ldw, sWo, sWi,
                                  C32, ldc32, sC32o, sC32i, C16, ldc16, sC16o, sC16i,
                                  bias, K, inner, relu, rowMul, rowOff);
}

static inline void gemm1(hipStream_t st, const _Float16* A, int lda,
                         const _Float16* W, int ldw, const float* bias, int relu,
                         float* C32, int ldc32, _Float16* C16, int ldc16,
                         int M, int N, int K)
{
  gemm(st, A, lda, 0, 0, W, ldw, 0, 0, bias, relu,
       C32, ldc32, 0, 0, C16, ldc16, 0, 0, M, N, K, 1, 1);
}

struct MhaScratch {
  _Float16 *Q16, *K16, *V16, *vT16, *probs16, *O16;
  float* scores;
};

static void run_mha(hipStream_t st, const MhaScratch& s,
                    const _Float16* q16, const _Float16* kv16, int Nq, int Nk,
                    const _Float16* inw, const float* inb,
                    const _Float16* outw, const float* outb,
                    float* o32, int ldo32, _Float16* o16, int ldo16)
{
  const int Mq = BB * Nq, Mk = BB * Nk;
  gemm1(st, q16, DD, inw, DD, inb, 0, nullptr, 0, s.Q16, DD, Mq, DD, DD);
  gemm1(st, kv16, DD, inw + DD * DD, DD, inb + DD, 0, nullptr, 0, s.K16, DD, Mk, DD, DD);
  gemm1(st, kv16, DD, inw + 2 * DD * DD, DD, inb + 2 * DD, 0, nullptr, 0, s.V16, DD, Mk, DD, DD);
  int tot = BB * NH * DH * Nk;
  cma_vT<<<(tot + 255) / 256, 256, 0, st>>>(s.V16, s.vT16, Nk);
  // scores[b,h] = Qh @ Kh^T
  gemm(st, s.Q16, DD, (long)Nq * DD, DH, s.K16, DD, (long)Nk * DD, DH,
       nullptr, 0,
       s.scores, Nk, (long)NH * Nq * Nk, (long)Nq * Nk,
       nullptr, 0, 0, 0, Nq, Nk, DH, BB, NH);
  cma_softmax<<<BB * NH * Nq, Nk, 0, st>>>(s.scores, s.probs16, Nk, 0.17677669529663687f);
  // O[b,h] = P @ V
  gemm(st, s.probs16, Nk, (long)NH * Nq * Nk, (long)Nq * Nk,
       s.vT16, Nk, (long)NH * DH * Nk, (long)DH * Nk,
       nullptr, 0,
       nullptr, 0, 0, 0,
       s.O16, DD, (long)Nq * DD, DH, Nq, DH, Nk, BB, NH);
  gemm1(st, s.O16, DD, outw, DD, outb, 0, o32, ldo32, o16, ldo16, Mq, DD, DD);
}

static void up_pass(hipStream_t st, const _Float16* x, int S,
                    const _Float16* We, const _Float16* Wo, const float* bias,
                    float* o32, _Float16* o16, _Float16* colbuf)
{
  int tot = BB * S * 256;
  cma_im2col_up<<<(tot + 255) / 256, 256, 0, st>>>(x, colbuf, S, 0, -1);
  gemm(st, colbuf, 256, 0, 0, We, 256, 0, 0, bias, 0,
       o32, DD, 0, 0, o16, DD, 0, 0, BB * S, DD, 256, 1, 1, 2, 0);
  cma_im2col_up<<<(tot + 255) / 256, 256, 0, st>>>(x, colbuf, S, 1, 0);
  gemm(st, colbuf, 256, 0, 0, Wo, 256, 0, 0, bias, 0,
       o32, DD, 0, 0, o16, DD, 0, 0, BB * S, DD, 256, 1, 1, 2, 1);
}

// ---------------------------------------------------------------------------

extern "C" void kernel_launch(void* const* d_in, const int* in_sizes, int n_in,
                              void* d_out, int out_size, void* d_ws, size_t ws_size,
                              hipStream_t stream)
{
  (void)in_sizes; (void)n_in; (void)out_size; (void)ws_size;
  auto F = [&](int i) { return (const float*)d_in[i]; };

  // input index map (recursive insertion order of setup_inputs dict)
  // 0 text, 1 image, 2..5 norms,
  // hier: mha[s]: 6+4s..9+4s (in_w,in_b,out_w,out_b) for s=0..2,
  //       down_w 18,19  down_b 20,21  up_w 22,23  up_b 24,25  level_w 26
  // cross: 27..30
  // relational: rel[r]: 31+6r..36+6r (w1,b1,w2,b2,w3,b3), agg 55..58
  // ap: 59 w1, 60 b1, 61 ln_w, 62 ln_b, 63 w2, 64 b2
  const int IW[4] = {6, 10, 14, 27}, IB[4] = {7, 11, 15, 28};
  const int OW[4] = {8, 12, 16, 29}, OB[4] = {9, 13, 17, 30};

  Arena ar{(char*)d_ws, 0};
  // f16 weights
  _Float16 *inw16[4], *outw16[4];
  for (int s = 0; s < 4; ++s) {
    inw16[s]  = (_Float16*)ar.take(3 * DD * DD * 2);
    outw16[s] = (_Float16*)ar.take(DD * DD * 2);
  }
  _Float16* dwT[2];  for (int i = 0; i < 2; ++i) dwT[i]  = (_Float16*)ar.take(128 * 384 * 2);
  _Float16* upWe[2]; for (int i = 0; i < 2; ++i) upWe[i] = (_Float16*)ar.take(128 * 256 * 2);
  _Float16* upWo[2]; for (int i = 0; i < 2; ++i) upWo[i] = (_Float16*)ar.take(128 * 256 * 2);
  _Float16* w1all = (_Float16*)ar.take(NREL * 128 * 256 * 2);
  _Float16* w2all = (_Float16*)ar.take(NREL * 64 * 128 * 2);
  float* b2all = (float*)ar.take(NREL * 64 * 4);
  float* w3all = (float*)ar.take(NREL * 64 * 4);
  float* b3all = (float*)ar.take(NREL * 4);
  _Float16* aggw1 = (_Float16*)ar.take(256 * 512 * 2);
  _Float16* aggw2 = (_Float16*)ar.take(128 * 256 * 2);
  _Float16* apw1  = (_Float16*)ar.take(256 * 384 * 2);
  _Float16* apw2  = (_Float16*)ar.take(128 * 256 * 2);
  // activations
  const size_t TOK = (size_t)BB * NN * DD;           // 131072
  _Float16* at16 = (_Float16*)ar.take(TOK * 2);
  _Float16* ai16 = (_Float16*)ar.take(TOK * 2);
  _Float16* qd1 = (_Float16*)ar.take(TOK);           // BB*128*128 halves
  _Float16* kd1 = (_Float16*)ar.take(TOK);
  _Float16* qd2 = (_Float16*)ar.take(TOK / 2);
  _Float16* kd2 = (_Float16*)ar.take(TOK / 2);
  _Float16* colbuf = (_Float16*)ar.take((size_t)BB * 128 * 384 * 2);
  MhaScratch ms;
  ms.Q16 = (_Float16*)ar.take(TOK * 2);
  ms.K16 = (_Float16*)ar.take(TOK * 2);
  ms.V16 = (_Float16*)ar.take(TOK * 2);
  ms.vT16 = (_Float16*)ar.take(TOK * 2);
  ms.probs16 = (_Float16*)ar.take((size_t)BB * NH * NN * NN * 2);
  ms.O16 = (_Float16*)ar.take(TOK * 2);
  ms.scores = (float*)ar.take((size_t)BB * NH * NN * NN * 4);
  float* a0  = (float*)ar.take(TOK * 4);
  float* a1u = (float*)ar.take(TOK * 4);
  float* a2u = (float*)ar.take(TOK * 4);
  _Float16* a1_16 = (_Float16*)ar.take(TOK);
  _Float16* a2_16 = (_Float16*)ar.take(TOK / 2);
  _Float16* t128  = (_Float16*)ar.take(TOK);
  _Float16* arel = (_Float16*)ar.take(NREL * TOK * 2);
  _Float16* crel = (_Float16*)ar.take(NREL * TOK * 2);
  _Float16* allr = (_Float16*)ar.take((size_t)BB * NN * 512 * 2);
  _Float16* aggh = (_Float16*)ar.take((size_t)BB * NN * 256 * 2);
  _Float16* comb = (_Float16*)ar.take((size_t)BB * NN * 384 * 2);
  _Float16* h2_16 = (_Float16*)ar.take((size_t)BB * NN * 256 * 2);
  float* hbuf  = (float*)ar.take((size_t)BB * NN * 256 * 4);
  float* projf = (float*)ar.take(TOK * 4);

  float* alT = (float*)d_out;            // (B,N,D) text slot
  float* alI = alT + TOK;                // image slot

  auto cvt = [&](const float* src, _Float16* dst, int n) {
    cma_cvt<<<(n + 255) / 256, 256, 0, stream>>>(src, dst, n);
  };

  // ---- weight prep ----
  for (int s = 0; s < 4; ++s) {
    cvt(F(IW[s]), inw16[s], 3 * DD * DD);
    cvt(F(OW[s]), outw16[s], DD * DD);
  }
  cma_pack_dw<<<(128 * 384 + 255) / 256, 256, 0, stream>>>(F(18), dwT[0]);
  cma_pack_dw<<<(128 * 384 + 255) / 256, 256, 0, stream>>>(F(19), dwT[1]);
  for (int i = 0; i < 2; ++i) {
    cma_pack_uw<<<(128 * 256 + 255) / 256, 256, 0, stream>>>(F(22 + i), upWe[i], 1, 3);
    cma_pack_uw<<<(128 * 256 + 255) / 256, 256, 0, stream>>>(F(22 + i), upWo[i], 0, 2);
  }
  for (int r = 0; r < NREL; ++r) {
    cvt(F(31 + 6 * r), w1all + (size_t)r * 128 * 256, 128 * 256);
    cvt(F(33 + 6 * r), w2all + (size_t)r * 64 * 128, 64 * 128);
    hipMemcpyAsync(b2all + r * 64, d_in[34 + 6 * r], 64 * 4, hipMemcpyDeviceToDevice, stream);
    hipMemcpyAsync(w3all + r * 64, d_in[35 + 6 * r], 64 * 4, hipMemcpyDeviceToDevice, stream);
    hipMemcpyAsync(b3all + r,      d_in[36 + 6 * r], 4,      hipMemcpyDeviceToDevice, stream);
  }
  cvt(F(55), aggw1, 256 * 512);
  cvt(F(57), aggw2, 128 * 256);
  cvt(F(59), apw1, 256 * 384);
  cvt(F(63), apw2, 128 * 256);

  // ---- initial layernorms ----
  cma_ln<<<BB * NN, 128, 0, stream>>>(F(0), F(2), F(3), alT, at16, DD);
  cma_ln<<<BB * NN, 128, 0, stream>>>(F(1), F(4), F(5), alI, ai16, DD);

  const int M1 = BB * NN;  // 1024

  for (int dir = 0; dir < 2; ++dir) {
    _Float16* m1 = dir ? ai16 : at16;
    _Float16* m2 = dir ? at16 : ai16;
    float* al1 = dir ? alI : alT;

    // ---- hierarchical: downsample chains ----
    int tot0 = BB * 128 * 384;
    cma_im2col_down<<<(tot0 + 255) / 256, 256, 0, stream>>>(m1, colbuf, 256);
    gemm1(stream, colbuf, 384, dwT[0], 384, F(20), 0, nullptr, 0, qd1, DD, BB * 128, DD, 384);
    cma_im2col_down<<<(tot0 + 255) / 256, 256, 0, stream>>>(m2, colbuf, 256);
    gemm1(stream, colbuf, 384, dwT[0], 384, F(20), 0, nullptr, 0, kd1, DD, BB * 128, DD, 384);
    int tot1 = BB * 64 * 384;
    cma_im2col_down<<<(tot1 + 255) / 256, 256, 0, stream>>>(qd1, colbuf, 128);
    gemm1(stream, colbuf, 384, dwT[1], 384, F(21), 0, nullptr, 0, qd2, DD, BB * 64, DD, 384);
    cma_im2col_down<<<(tot1 + 255) / 256, 256, 0, stream>>>(kd1, colbuf, 128);
    gemm1(stream, colbuf, 384, dwT[1], 384, F(21), 0, nullptr, 0, kd2, DD, BB * 64, DD, 384);

    // ---- per-level MHA ----
    run_mha(stream, ms, m1, m2, 256, 256, inw16[0], F(IB[0]), outw16[0], F(OB[0]),
            a0, DD, nullptr, 0);
    run_mha(stream, ms, qd1, kd1, 128, 128, inw16[1], F(IB[1]), outw16[1], F(OB[1]),
            nullptr, 0, a1_16, DD);
    run_mha(stream, ms, qd2, kd2, 64, 64, inw16[2], F(IB[2]), outw16[2], F(OB[2]),
            nullptr, 0, a2_16, DD);

    // ---- upsample + level mix ----
    up_pass(stream, a1_16, 128, upWe[0], upWo[0], F(24), a1u, nullptr, colbuf);
    up_pass(stream, a2_16, 64, upWe[0], upWo[0], F(24), nullptr, t128, colbuf);
    up_pass(stream, t128, 128, upWe[1], upWo[1], F(25), a2u, nullptr, colbuf);
    cma_wsum<<<(int)((TOK + 255) / 256), 256, 0, stream>>>(a0, a1u, a2u, F(26), comb);

    // ---- standard cross attention -> comb[:,128:256] ----
    run_mha(stream, ms, m1, m2, 256, 256, inw16[3], F(IB[3]), outw16[3], F(OB[3]),
            nullptr, 0, comb + 128, 384);

    // ---- relational -> comb[:,256:384] ----
    for (int r = 0; r < NREL; ++r) {
      const _Float16* w1r = w1all + (size_t)r * 128 * 256;
      gemm1(stream, m1, DD, w1r, 256, nullptr, 0,
            nullptr, 0, arel + (size_t)r * TOK, DD, M1, DD, DD);
      gemm1(stream, m2, DD, w1r + 128, 256, F(32 + 6 * r), 0,
            nullptr, 0, crel + (size_t)r * TOK, DD, M1, DD, DD);
    }
    cma_rel<<<dim3(BB * NN, NREL), 256, 0, stream>>>(arel, crel, m2, w2all,
                                                     b2all, w3all, b3all, allr);
    gemm1(stream, allr, 512, aggw1, 512, F(56), 1, nullptr, 0, aggh, 256, M1, 256, 512);
    gemm1(stream, aggh, 256, aggw2, 256, F(58), 0, nullptr, 0, comb + 256, 384, M1, DD, 256);

    // ---- adaptive projection + residual ----
    gemm1(stream, comb, 384, apw1, 384, F(60), 1, hbuf, 256, nullptr, 0, M1, 256, 384);
    cma_ln<<<BB * NN, 128, 0, stream>>>(hbuf, F(61), F(62), nullptr, h2_16, 256);
    gemm1(stream, h2_16, 256, apw2, 256, F(64), 0, projf, DD, nullptr, 0, M1, DD, 256);
    cma_resid<<<(int)((TOK + 255) / 256), 256, 0, stream>>>(al1, projf, m1, (int)TOK);
  }
}
